// LEAF_57681410786028
// MI455X (gfx1250) — matrix operand order repaired and checked by reading, not compiled, not run on hardware
//
#include <hip/hip_runtime.h>
#include <hip/hip_fp16.h>

#define N_BINS      40
#define FILTER_SIZE 401
#define POOL_SIZE   401
#define STRIDE      160
#define BATCH       16
#define N_SAMPLES   64000
#define N_POOL      400
#define EPS         1e-6f
#define SQRT_2PI    2.5066282746310002f

#define KBLOCKS 13            // ceil(401/32): K padded to 416
#define MTILES  5             // 5 M-tiles of 16 rows: [8 cos bins | 8 sin bins]
#define XPAD_L  320
#define XPLEN   64768         // padded f16 signal length per batch
#define NP      8             // pooled outputs per block
#define NCHUNK  (N_POOL / NP) // 50
#define TTILES  96            // 16-wide modulus time tiles per chunk (covers (NP-1)*160+401)

typedef __attribute__((ext_vector_type(16))) _Float16 v16h;
typedef __attribute__((ext_vector_type(8)))  float    v8f;

struct __attribute__((packed)) v16h_pack { v16h v; };

__device__ __forceinline__ int fdiv_floor(int a, int b) {
    int q = a / b, rm = a % b;
    return q - ((rm != 0) && ((rm < 0) != (b < 0)));
}

// ---------------- setup: pad + f16-convert input signal ----------------
__global__ void k_pad(const float* __restrict__ x, _Float16* __restrict__ xp) {
    int i = blockIdx.x * blockDim.x + threadIdx.x;
    if (i >= BATCH * XPLEN) return;
    int b = i / XPLEN, j = i % XPLEN;
    int s = j - XPAD_L;
    float v = (s >= 0 && s < N_SAMPLES) ? x[b * N_SAMPLES + s] : 0.0f;
    xp[i] = (_Float16)v;
}

// ---------------- setup: Gabor filters pre-permuted into WMMA A-fragment layout ----
// frag[((g*KBLOCKS + kb)*32 + lane)*16 + e], in contiguous-k 16-bit A 16x32 order (not the ISA table order):
//   lane: m = lane&15, half = lane>>4
//   element e: klocal = 16*half + e
__global__ void k_filt(const float* __restrict__ mu, const float* __restrict__ sigma,
                       _Float16* __restrict__ frag) {
    int i = blockIdx.x * blockDim.x + threadIdx.x;
    if (i >= MTILES * KBLOCKS * 512) return;
    int e    = i & 15;
    int lane = (i >> 4) & 31;
    int kb   = (i >> 9) % KBLOCKS;
    int g    = i / (KBLOCKS * 512);
    int half = lane >> 4;
    int m    = lane & 15;
    int p    = e >> 1;
    int klocal = 16 * half + e;
    int k = kb * 32 + klocal;
    float val = 0.0f;
    if (k < FILTER_SIZE) {
        int bin = g * 8 + (m & 7);
        float tf = (float)k - 200.0f;
        float s  = sigma[bin];
        float gs = expf(-tf * tf / (2.0f * s * s)) / (SQRT_2PI * s);
        float ph = mu[bin] * tf;
        val = ((m < 8) ? cosf(ph) : sinf(ph)) * gs;
    }
    frag[i] = (_Float16)val;
}

// ---------------- setup: Gaussian lowpass table [bin][tap] ----------------
__global__ void k_lp(const float* __restrict__ sigma_lp, float* __restrict__ lp) {
    int i = blockIdx.x * blockDim.x + threadIdx.x;
    if (i >= N_BINS * POOL_SIZE) return;
    int bin = i / POOL_SIZE, k = i % POOL_SIZE;
    float tf = (float)k - 200.0f;
    float denom = sigma_lp[bin] * 0.5f * (float)(POOL_SIZE - 1);
    float z = tf / denom;
    lp[i] = expf(-0.5f * z * z);
}

// ---------------- main: WMMA Gabor conv + modulus + fused pooling ----------------
__global__ __launch_bounds__(128) void k_main(const _Float16* __restrict__ xp,
                                              const _Float16* __restrict__ frag,
                                              const float* __restrict__ lp,
                                              float* __restrict__ pooled) {
    __shared__ float lp_s[N_BINS * POOL_SIZE];   // 64,160 B
    __shared__ float pool_s[N_BINS * NP];        //  1,280 B

    const int tid = threadIdx.x;
    const int b   = blockIdx.y;
    const int p0  = blockIdx.x * NP;
    const int chunk_t0 = p0 * STRIDE - 120;      // SAME pool padding: pad_left = 120

    for (int i = tid; i < N_BINS * POOL_SIZE; i += 128) lp_s[i] = lp[i];
    if (tid < N_BINS * NP) pool_s[tid] = 0.0f;
    __syncthreads();

    const int wave = tid >> 5;
    const int lane = tid & 31;
    const int nn   = lane & 15;
    const int half = lane >> 4;
    const _Float16* xb = xp + (size_t)b * XPLEN;

    for (int tile = wave; tile < TTILES; tile += 4) {
        const int t0   = chunk_t0 + tile * 16;
        // xp index of x[t + k - 200] is t + k + 120 (XPAD_L=320, conv pad_left=200)
        const _Float16* xrow = xb + (t0 + 120 + nn + 16 * half);
        __builtin_prefetch(xrow + 1024, 0, 3);

        // Issue ALL B-fragment loads up front (26 x b128 in flight) so the
        // memory pipeline runs in parallel and the WMMA stream needs at most
        // staircase loadcnt waits instead of 13 full-latency stalls.
        v16h bf[KBLOCKS];
#pragma unroll
        for (int kb = 0; kb < KBLOCKS; ++kb)
            bf[kb] = ((const v16h_pack*)(xrow + 32 * kb))->v;

        v8f acc[MTILES] = {};
#pragma unroll
        for (int kb = 0; kb < KBLOCKS; ++kb) {
#pragma unroll
            for (int g = 0; g < MTILES; ++g) {
                const v16h afrag =
                    *(const v16h*)(frag + ((size_t)((g * KBLOCKS + kb) * 32 + lane) << 4));
                acc[g] = __builtin_amdgcn_wmma_f32_16x16x32_f16(
                    false, afrag, false, bf[kb], (short)0, acc[g], false, false);
            }
        }

        // modulus + pooling accumulation. C layout: lanes 0-15 -> rows 0-7 (cos),
        // lanes 16-31 -> rows 8-15 (sin) of same bins; shfl_xor(16) pairs them.
        const int t = t0 + nn;
        if ((unsigned)t < (unsigned)N_SAMPLES) {
            const int p_first = fdiv_floor(t - 121, STRIDE);  // ceil((t-280)/160)
#pragma unroll
            for (int g = 0; g < MTILES; ++g) {
#pragma unroll
                for (int r = 0; r < 8; ++r) {
                    float v  = acc[g][r];
                    float pv = __shfl_xor(v, 16, 32);
                    float md = v * v + pv * pv;
                    int bin  = g * 8 + r;
#pragma unroll
                    for (int dp = 0; dp < 2; ++dp) {
                        int p   = p_first + dp + 2 * half;  // halves split p-candidates
                        int klp = t - p * STRIDE + 120;
                        if (p >= p0 && p < p0 + NP && klp >= 0 && klp < POOL_SIZE) {
                            atomicAdd(&pool_s[bin * NP + (p - p0)],
                                      md * lp_s[bin * POOL_SIZE + klp]);
                        }
                    }
                }
            }
        }
    }

    __syncthreads();
    if (tid < N_BINS * NP) {
        int bin = tid / NP, pp = tid % NP;
        pooled[((size_t)b * N_BINS + bin) * N_POOL + p0 + pp] = pool_s[tid];
    }
}

// ---------------- epilogue: relu + EMA scan + PCEN, in-place on d_out ----------------
__global__ void k_post(float* __restrict__ out, const float* __restrict__ w_smooth,
                       const float* __restrict__ alpha, const float* __restrict__ delta,
                       const float* __restrict__ root) {
    int id = blockIdx.x * blockDim.x + threadIdx.x;
    if (id >= BATCH * N_BINS) return;
    int b = id / N_BINS, bin = id % N_BINS;
    float w  = fminf(fmaxf(w_smooth[bin], 0.0f), 1.0f);
    float a  = alpha[bin], d = delta[bin], r = root[bin];
    float dr = powf(d, r);
    float* row = out + ((size_t)b * N_BINS + bin) * N_POOL;
    float M = 0.0f;
    for (int p = 0; p < N_POOL; ++p) {
        float pv = fmaxf(row[p], 0.0f);
        M = (p == 0) ? pv : (w * pv + (1.0f - w) * M);
        row[p] = powf(pv / powf(EPS + M, a) + d, r) - dr;
    }
}

extern "C" void kernel_launch(void* const* d_in, const int* in_sizes, int n_in,
                              void* d_out, int out_size, void* d_ws, size_t ws_size,
                              hipStream_t stream) {
    const float* x        = (const float*)d_in[0];
    const float* mu       = (const float*)d_in[1];
    const float* sigma    = (const float*)d_in[2];
    const float* sigma_lp = (const float*)d_in[3];
    const float* w_smooth = (const float*)d_in[4];
    const float* alpha    = (const float*)d_in[5];
    const float* delta    = (const float*)d_in[6];
    const float* root     = (const float*)d_in[7];

    char* ws = (char*)d_ws;
    _Float16* xp   = (_Float16*)ws;                                      // 2,072,576 B
    _Float16* frag = (_Float16*)(ws + (size_t)BATCH * XPLEN * 2);        //    66,560 B
    float*    lp   = (float*)(ws + (size_t)BATCH * XPLEN * 2
                                 + (size_t)MTILES * KBLOCKS * 512 * 2);  //    64,160 B
    float* out = (float*)d_out;

    { int n = BATCH * XPLEN;          k_pad <<<(n + 255) / 256, 256, 0, stream>>>(x, xp); }
    { int n = MTILES * KBLOCKS * 512; k_filt<<<(n + 255) / 256, 256, 0, stream>>>(mu, sigma, frag); }
    { int n = N_BINS * POOL_SIZE;     k_lp  <<<(n + 255) / 256, 256, 0, stream>>>(sigma_lp, lp); }

    k_main<<<dim3(NCHUNK, BATCH), 128, 0, stream>>>(xp, frag, lp, out);

    { int n = BATCH * N_BINS;         k_post<<<(n + 127) / 128, 128, 0, stream>>>(out, w_smooth, alpha, delta, root); }
}
